// OMAD_PriorNet_3204045603269
// MI455X (gfx1250) — matrix operand order repaired and checked by reading, not compiled, not run on hardware
//
#include <hip/hip_runtime.h>

typedef __attribute__((ext_vector_type(16))) _Float16 v16h;
typedef __attribute__((ext_vector_type(8)))  _Float16 v8h;
typedef __attribute__((ext_vector_type(8)))  float    v8f;

#define BB 8
#define NN 32768
#define PP (BB*NN)
#define MM 16
#define EPS 1e-5f
#define PT 4   // point-tiles (16 rows each) per wave: 4 accumulators

static __device__ __forceinline__ void atomicMaxF(float* a, float v) {
  if (v >= 0.0f) atomicMax((int*)a, __float_as_int(v));
  else atomicMin((unsigned int*)a, (unsigned int)__float_as_int(v));
}

// ---------------- tiny utility kernels ----------------
__global__ void k_zero(float* p, int n) {
  int i = blockIdx.x * 256 + threadIdx.x;
  if (i < n) p[i] = 0.0f;
}
__global__ void k_fill(float* p, int n, float v) {
  int i = blockIdx.x * 256 + threadIdx.x;
  if (i < n) p[i] = v;
}

// ---------------- stage A: cluster assignment + means ----------------
__global__ void k_assign(const float* __restrict__ x, const float* __restrict__ node,
                         int* __restrict__ mi, float* __restrict__ csum, float* __restrict__ ccnt) {
  int p = blockIdx.x * 256 + threadIdx.x;
  if (p >= PP) return;
  int b = p >> 15;
  float px = x[3*p+0], py = x[3*p+1], pz = x[3*p+2];
  int best = 0; float bd = 3.4e38f;
  for (int m = 0; m < MM; ++m) {
    float dx = px - node[(b*MM+m)*3+0];
    float dy = py - node[(b*MM+m)*3+1];
    float dz = pz - node[(b*MM+m)*3+2];
    float d = dx*dx + dy*dy + dz*dz;
    if (d < bd) { bd = d; best = m; }
  }
  mi[p] = best;
  atomicAdd(&csum[(b*MM+best)*3+0], px);
  atomicAdd(&csum[(b*MM+best)*3+1], py);
  atomicAdd(&csum[(b*MM+best)*3+2], pz);
  atomicAdd(&ccnt[b*MM+best], 1.0f);
}

// q layout: [b][d][m]  (stride 48 per batch)
__global__ void k_cluster_fin(const float* __restrict__ csum, const float* __restrict__ ccnt,
                              float* __restrict__ q) {
  int t = threadIdx.x;      // 128 threads: (b,m)
  if (t >= BB*MM) return;
  int b = t / MM, m = t % MM;
  float inv = 1.0f / (ccnt[t] + 1e-5f);
  for (int d = 0; d < 3; ++d) q[b*48 + d*16 + m] = csum[t*3+d] * inv;
}

// f1 input: [P][32] f16, ch0-2 = x - center, ch3-5 = sn, rest 0
__global__ void k_build_f1(const float* __restrict__ x, const float* __restrict__ sn,
                           const float* __restrict__ q, const int* __restrict__ mi,
                           _Float16* __restrict__ dst) {
  int p = blockIdx.x * 256 + threadIdx.x;
  if (p >= PP) return;
  int b = p >> 15, m = mi[p];
  _Float16* r = dst + (size_t)p * 32;
  for (int d = 0; d < 3; ++d) r[d]   = (_Float16)(x[3*p+d] - q[b*48 + d*16 + m]);
  for (int d = 0; d < 3; ++d) r[3+d] = (_Float16)sn[3*p+d];
  for (int c = 6; c < 32; ++c) r[c] = (_Float16)0.0f;
}

// inst input: [P][32] f16, ch0-2 = x, ch3-5 = sn
__global__ void k_build_hin(const float* __restrict__ x, const float* __restrict__ sn,
                            _Float16* __restrict__ dst) {
  int p = blockIdx.x * 256 + threadIdx.x;
  if (p >= PP) return;
  _Float16* r = dst + (size_t)p * 32;
  for (int d = 0; d < 3; ++d) r[d]   = (_Float16)x[3*p+d];
  for (int d = 0; d < 3; ++d) r[3+d] = (_Float16)sn[3*p+d];
  for (int c = 6; c < 32; ++c) r[c] = (_Float16)0.0f;
}

// ---------------- WMMA weight packing ----------------
// B-fragment layout (V_WMMA_F32_16X16X32_F16, 32x16 f16 B):
//   lane L holds column n = L%16; element i holds K = kt*32 + (i<8 ? i : i+8) + 8*(L>=16)
// Wpack[((kt*CT + ct)*32 + lane)*16 + i] = W[n][k]  (0 if out of range)
__global__ void k_pack_w(const float* __restrict__ W, _Float16* __restrict__ Wp,
                         int Cin, int Cout, int CT) {
  int lane = threadIdx.x;              // 32 threads
  int tile = blockIdx.x;               // KT*CT blocks
  int kt = tile / CT, ct = tile % CT;
  int n  = ct*16 + (lane & 15);
  int hi = lane >> 4;
  for (int i = 0; i < 16; ++i) {
    int k = kt*32 + (i < 8 ? i : i + 8) + hi*8;
    float v = (n < Cout && k < Cin) ? W[(size_t)n * Cin + k] : 0.0f;
    Wp[((size_t)tile*32 + lane)*16 + i] = (_Float16)v;
  }
}

// ---------------- the WMMA conv1x1 GEMM ----------------
// X: [P][Cinp] f16 (Cinp mult of 32), Y: [P][Coutp] f16 (Coutp mult of 16)
// Each wave: PT accumulator tiles (PT*16 points x 16 cout), full K reduction.
// Per K-step: 1 B-fragment load (reused) -> PT A-loads -> PT WMMAs.
// A 16-bit layout: lane L row = pt, elem i -> K = (i<8 ? i : i+8) + 8*(L>=16)
__global__ void __launch_bounds__(256)
k_wmma_conv(const _Float16* __restrict__ X, const _Float16* __restrict__ Wp,
            const float* __restrict__ bias, int CoutReal,
            _Float16* __restrict__ Y, int Cinp, int Coutp) {
  const int lane = threadIdx.x & 31;
  const int wave = threadIdx.x >> 5;
  const int CT = Coutp >> 4;
  const int KT = Cinp >> 5;
  int ctile, ptgrp, grpPerBlock;
  if (CT >= 8) { ctile = wave; ptgrp = 0; grpPerBlock = 1; }
  else { ctile = wave % CT; ptgrp = wave / CT; grpPerBlock = 8 / CT; }
  const int ptsPerBlock = grpPerBlock * PT * 16;
  const int ptbase = blockIdx.x * ptsPerBlock + ptgrp * PT * 16;
  if (ptbase >= PP) return;                      // block-uniform
  const int hi = lane >> 4;

  const _Float16* xrow[PT];
#pragma unroll
  for (int t = 0; t < PT; ++t)
    xrow[t] = X + (size_t)(ptbase + t*16 + (lane & 15)) * Cinp + hi * 8;

  v8f c[PT] = {};
  for (int kt = 0; kt < KT; ++kt) {
    // weight fragment: loaded once, reused by all PT WMMAs this step
    v16h bf = *(const v16h*)(Wp + (((size_t)(kt*CT + ctile)*32 + lane) * 16));
#pragma unroll
    for (int t = 0; t < PT; ++t) {
      v8h a0 = *(const v8h*)(xrow[t] + kt*32);        // K = kt*32 + 8*hi + [0..7]
      v8h a1 = *(const v8h*)(xrow[t] + kt*32 + 16);   // K = kt*32 + 16 + 8*hi + [0..7]
      v16h a;
#pragma unroll
      for (int i = 0; i < 8; ++i) { a[i] = a0[i]; a[i+8] = a1[i]; }
      c[t] = __builtin_amdgcn_wmma_f32_16x16x32_f16(false, a, false, bf,
                                                    (short)0, c[t], false, false);
    }
  }
  const int col = ctile*16 + (lane & 15);
  float bv = (bias && col < CoutReal) ? bias[col] : 0.0f;
#pragma unroll
  for (int t = 0; t < PT; ++t) {
#pragma unroll
    for (int j = 0; j < 8; ++j) {                  // D: vgpr j -> M = j + 8*hi
      int m = j + hi*8;
      Y[(size_t)(ptbase + t*16 + m) * Coutp + col] = (_Float16)(c[t][j] + bv);
    }
  }
}

// ---------------- batch-norm stats + apply ----------------
#define NRG 512
__global__ void k_colstats(const _Float16* __restrict__ Z, float* __restrict__ sum,
                           float* __restrict__ sumsq, int Cp) {
  int t = blockIdx.x * 256 + threadIdx.x;        // Cp*NRG threads
  int c = t % Cp, rg = t / Cp;
  if (rg >= NRG) return;
  float s = 0.0f, s2 = 0.0f;
  for (int r = rg; r < PP; r += NRG) {
    float v = (float)Z[(size_t)r * Cp + c];
    s += v; s2 += v*v;
  }
  atomicAdd(&sum[c], s); atomicAdd(&sumsq[c], s2);
}
__global__ void k_stats_fin(const float* __restrict__ sum, const float* __restrict__ sumsq,
                            const float* __restrict__ g, const float* __restrict__ beta,
                            float* __restrict__ scale, float* __restrict__ shift, int C) {
  int c = threadIdx.x; if (c >= C) return;
  float invP = 1.0f / (float)PP;
  float mean = sum[c] * invP;
  float var  = sumsq[c] * invP - mean*mean;
  float sc   = g[c] * rsqrtf(var + EPS);
  scale[c] = sc; shift[c] = beta[c] - sc*mean;
}
__global__ void k_normrelu(_Float16* __restrict__ Z, const float* __restrict__ scale,
                           const float* __restrict__ shift, size_t n, int cmask) {
  size_t i = (size_t)blockIdx.x * 256 + threadIdx.x;
  size_t stride = (size_t)gridDim.x * 256;
  for (; i < n; i += stride) {
    int c = (int)(i & (size_t)cmask);
    float v = (float)Z[i];
    v = fmaf(scale[c], v, shift[c]);
    Z[i] = (_Float16)fmaxf(v, 0.0f);
  }
}

// ---------------- cluster ops ----------------
__global__ void k_clustermax(const _Float16* __restrict__ Z, const int* __restrict__ mi,
                             float* __restrict__ fm, int Cp) {
  int p = blockIdx.x * 256 + threadIdx.x;
  if (p >= PP) return;
  int b = p >> 15, m = mi[p];
  const _Float16* zr = Z + (size_t)p * Cp;
  float* fr = fm + (size_t)(b*MM + m) * Cp;
  for (int c = 0; c < Cp; ++c) atomicMaxF(&fr[c], (float)zr[c]);
}
__global__ void k_build_f2(const _Float16* __restrict__ f1, const float* __restrict__ fm1,
                           const int* __restrict__ mi, _Float16* __restrict__ dst) {
  int p = blockIdx.x * 256 + threadIdx.x;
  if (p >= PP) return;
  int b = p >> 15, m = mi[p];
  const _Float16* s = f1 + (size_t)p * 64;
  const float* f = fm1 + (size_t)(b*MM + m) * 64;
  _Float16* r = dst + (size_t)p * 128;
  for (int c = 0; c < 64; ++c) r[c]    = s[c];
  for (int c = 0; c < 64; ++c) r[64+c] = (_Float16)f[c];
}

// ---------------- small (M=16) graph, scalar f32 ----------------
__global__ void k_knn(const float* __restrict__ q, int* __restrict__ knn) {
  int t = threadIdx.x; if (t >= BB*MM) return;   // 128 threads
  int b = t / MM, mq = t % MM;
  float d2[MM];
  for (int m = 0; m < MM; ++m) {
    float s = 0.0f;
    for (int d = 0; d < 3; ++d) {
      float df = q[b*48 + d*16 + mq] - q[b*48 + d*16 + m];
      s += df*df;
    }
    d2[m] = s;
  }
  int used = 0; // top-3 smallest, index tiebreak
  for (int k = 0; k < 3; ++k) {
    int bi = -1; float bv = 3.4e38f;
    for (int m = 0; m < MM; ++m) {
      if (used & (1 << m)) continue;
      if (d2[m] < bv) { bv = d2[m]; bi = m; }
    }
    used |= (1 << bi);
    knn[t*3 + k] = bi;
  }
}
__global__ void k_build_g(const float* __restrict__ q, const float* __restrict__ fm2,
                          const int* __restrict__ knn, float* __restrict__ Xg) {
  int pos = threadIdx.x + blockIdx.x * 256;      // 384 = b*48 + mq*3 + k
  if (pos >= 384) return;
  int b = pos / 48, rem = pos % 48, mq = rem / 3, k = rem % 3;
  int mi = knn[(b*MM + mq)*3 + k];
  float* r = Xg + (size_t)pos * 131;
  for (int d = 0; d < 3; ++d) r[d] = q[b*48 + d*16 + mi] - q[b*48 + d*16 + mq];
  const float* f = fm2 + (size_t)(b*MM + mi) * 128;
  for (int c = 0; c < 128; ++c) r[3+c] = f[c];
}
__global__ void k_sgemm(const float* __restrict__ X, const float* __restrict__ W,
                        const float* __restrict__ bias, float* __restrict__ Y,
                        int npos, int Cin, int Cout) {
  int t = blockIdx.x * 256 + threadIdx.x;
  if (t >= npos * Cout) return;
  int pos = t / Cout, o = t % Cout;
  const float* xr = X + (size_t)pos * Cin;
  const float* wr = W + (size_t)o * Cin;
  float acc = bias ? bias[o] : 0.0f;
  for (int c = 0; c < Cin; ++c) acc = fmaf(xr[c], wr[c], acc);
  Y[t] = acc;
}
__global__ void k_sbn(float* __restrict__ Y, const float* __restrict__ g,
                      const float* __restrict__ beta, int npos, int Cout) {
  int c = blockIdx.x;                            // one block per channel
  __shared__ float ss[256], ss2[256];
  float s = 0.0f, s2 = 0.0f;
  for (int p = threadIdx.x; p < npos; p += 256) {
    float v = Y[(size_t)p * Cout + c]; s += v; s2 += v*v;
  }
  ss[threadIdx.x] = s; ss2[threadIdx.x] = s2; __syncthreads();
  for (int o = 128; o > 0; o >>= 1) {
    if (threadIdx.x < o) { ss[threadIdx.x] += ss[threadIdx.x+o]; ss2[threadIdx.x] += ss2[threadIdx.x+o]; }
    __syncthreads();
  }
  float mean = ss[0] / npos;
  float var  = ss2[0] / npos - mean*mean;
  float sc = g[c] * rsqrtf(var + EPS);
  float sh = beta[c] - sc*mean;
  for (int p = threadIdx.x; p < npos; p += 256) {
    float v = fmaf(sc, Y[(size_t)p * Cout + c], sh);
    Y[(size_t)p * Cout + c] = fmaxf(v, 0.0f);
  }
}
__global__ void k_gmax(const float* __restrict__ G, float* __restrict__ gm) {
  int t = blockIdx.x * 256 + threadIdx.x;        // 128*256
  if (t >= 128*256) return;
  int pm = t / 256, c = t % 256;
  float v = G[(size_t)(pm*3+0)*256 + c];
  v = fmaxf(v, G[(size_t)(pm*3+1)*256 + c]);
  v = fmaxf(v, G[(size_t)(pm*3+2)*256 + c]);
  gm[t] = v;
}
__global__ void k_cat512(const float* __restrict__ G, const float* __restrict__ gm,
                         float* __restrict__ out) {
  int t = blockIdx.x * 256 + threadIdx.x;        // 384*512
  if (t >= 384*512) return;
  int pos = t / 512, c = t % 512;
  out[t] = (c < 256) ? G[(size_t)pos*256 + c] : gm[(size_t)(pos/3)*256 + (c-256)];
}
__global__ void k_knnfeat(const float* __restrict__ G, float* __restrict__ kf) {
  int t = blockIdx.x * 256 + threadIdx.x;        // 128*512
  if (t >= 128*512) return;
  int pm = t / 512, c = t % 512;
  float v = G[(size_t)(pm*3+0)*512 + c];
  v = fmaxf(v, G[(size_t)(pm*3+1)*512 + c]);
  v = fmaxf(v, G[(size_t)(pm*3+2)*512 + c]);
  kf[t] = v;
}
__global__ void k_build_nf(const float* __restrict__ fm2, const float* __restrict__ kf,
                           float* __restrict__ nf) {
  int t = blockIdx.x * 256 + threadIdx.x;        // 128*640
  if (t >= 128*640) return;
  int pm = t / 640, c = t % 640;
  nf[t] = (c < 128) ? fm2[(size_t)pm*128 + c] : kf[(size_t)pm*512 + (c-128)];
}
__global__ void k_part_nodes(const float* __restrict__ ko, const float* __restrict__ q,
                             float* __restrict__ out) {
  int t = threadIdx.x + blockIdx.x * 256;        // 384
  if (t >= 384) return;
  int pm = t / 3, d = t % 3, b = pm / MM, m = pm % MM;
  out[t] = ko[pm*4 + d] + q[b*48 + d*16 + m];
}
__global__ void k_coefmax(const _Float16* __restrict__ Z, float* __restrict__ coefs) {
  int p = blockIdx.x * 256 + threadIdx.x;
  if (p >= PP) return;
  int b = p >> 15;
  const _Float16* zr = Z + (size_t)p * 16;
  for (int c = 0; c < 10; ++c) atomicMaxF(&coefs[b*10 + c], (float)zr[c]);
}
__global__ void k_jp_out(const float* __restrict__ J, float* __restrict__ out) {
  int t = threadIdx.x; if (t >= 48) return;      // B*6
  int b = t / 6, c = t % 6;
  if (c < 3) out[464 + b*3 + c]       = J[b*6 + c];
  else       out[488 + b*3 + (c-3)]   = J[b*6 + c];
}

// ---------------- host orchestration ----------------
extern "C" void kernel_launch(void* const* d_in, const int* in_sizes, int n_in,
                              void* d_out, int out_size, void* d_ws, size_t ws_size,
                              hipStream_t stream) {
  (void)in_sizes; (void)n_in; (void)out_size; (void)ws_size;
  const float* x    = (const float*)d_in[0];
  const float* sn   = (const float*)d_in[1];
  const float* node = (const float*)d_in[2];
  // params flattened per layer: W, b, g, beta  (19 layers, insertion order)
  auto PW = [&](int L){ return (const float*)d_in[3 + 4*L + 0]; };
  auto Pb = [&](int L){ return (const float*)d_in[3 + 4*L + 1]; };
  auto Pg = [&](int L){ return (const float*)d_in[3 + 4*L + 2]; };
  auto Pz = [&](int L){ return (const float*)d_in[3 + 4*L + 3]; };
  float* out = (float*)d_out;

  // workspace carve-up (256B aligned)
  char* base = (char*)d_ws;
  size_t off = 0;
  auto alloc = [&](size_t sz) -> void* {
    void* p = base + off;
    off = (off + sz + 255) & ~(size_t)255;
    return p;
  };
  _Float16* bufA  = (_Float16*)alloc((size_t)PP * 128 * 2);
  _Float16* bufB  = (_Float16*)alloc((size_t)PP * 128 * 2);
  int*      minid = (int*)     alloc((size_t)PP * 4);
  float*    csum  = (float*)   alloc(BB*MM*3 * 4);
  float*    ccnt  = (float*)   alloc(BB*MM * 4);
  float*    q     = (float*)   alloc(BB*48 * 4);
  float*    fm1   = (float*)   alloc(BB*MM*64 * 4);
  float*    fm2   = (float*)   alloc(BB*MM*128 * 4);
  _Float16* wpack[8];
  for (int i = 0; i < 8; ++i) wpack[i] = (_Float16*)alloc(128*128*2);
  float*    stats = (float*)   alloc(256 * 4);   // sum[128], sumsq[128]
  float*    scsh  = (float*)   alloc(256 * 4);   // scale[128], shift[128]
  int*      knn   = (int*)     alloc(BB*MM*3 * 4);
  float*    Xg    = (float*)   alloc((size_t)384*131 * 4);
  float*    G1    = (float*)   alloc((size_t)384*256 * 4);
  float*    G2    = (float*)   alloc((size_t)384*256 * 4);
  float*    gm    = (float*)   alloc((size_t)128*256 * 4);
  float*    C5a   = (float*)   alloc((size_t)384*512 * 4);
  float*    C5b   = (float*)   alloc((size_t)384*512 * 4);
  float*    kf    = (float*)   alloc((size_t)128*512 * 4);
  float*    nf    = (float*)   alloc((size_t)128*640 * 4);
  float*    y512  = (float*)   alloc((size_t)128*512 * 4);
  float*    y256  = (float*)   alloc((size_t)128*256 * 4);
  float*    ko    = (float*)   alloc((size_t)128*4 * 4);
  float*    J1    = (float*)   alloc(BB*64 * 4);
  float*    J2    = (float*)   alloc(BB*64 * 4);
  float*    J3    = (float*)   alloc(BB*6 * 4);

  const int PB = PP / 256;   // 1024 blocks for per-point kernels

  // pack WMMA weights: slot -> (layer, Cin, Cout, Cinp, Coutp)
  struct { int L, Cin, Cout, Cinp, Coutp; } wl[8] = {
    {0, 6, 64, 32, 64}, {1, 64, 64, 64, 64}, {2, 64, 64, 64, 64},
    {3, 128, 128, 128, 128}, {4, 128, 128, 128, 128},
    {13, 6, 64, 32, 64}, {14, 64, 128, 64, 128}, {15, 128, 10, 128, 16}
  };
  for (int i = 0; i < 8; ++i) {
    int KT = wl[i].Cinp / 32, CT = wl[i].Coutp / 16;
    k_pack_w<<<KT*CT, 32, 0, stream>>>(PW(wl[i].L), wpack[i], wl[i].Cin, wl[i].Cout, CT);
  }

  auto conv = [&](int slot, const _Float16* X, _Float16* Y, const float* bias, int CoutReal) {
    int Cinp = wl[slot].Cinp, Coutp = wl[slot].Coutp;
    int CT = Coutp / 16;
    int ptsPerBlock = (CT >= 8 ? 1 : 8 / CT) * PT * 16;
    k_wmma_conv<<<PP / ptsPerBlock, 256, 0, stream>>>(X, wpack[slot], bias, CoutReal, Y, Cinp, Coutp);
  };
  auto bnrelu = [&](int L, _Float16* Z, int Cp) {
    k_zero<<<1, 256, 0, stream>>>(stats, 256);
    k_colstats<<<(Cp*NRG)/256, 256, 0, stream>>>(Z, stats, stats + 128, Cp);
    k_stats_fin<<<1, 128, 0, stream>>>(stats, stats + 128, Pg(L), Pz(L), scsh, scsh + 128, Cp);
    k_normrelu<<<8192, 256, 0, stream>>>(Z, scsh, scsh + 128, (size_t)PP * Cp, Cp - 1);
  };

  // Stage A: clustering
  k_zero<<<(BB*MM*3 + 255)/256, 256, 0, stream>>>(csum, BB*MM*3);
  k_zero<<<1, 256, 0, stream>>>(ccnt, BB*MM);
  k_assign<<<PB, 256, 0, stream>>>(x, node, minid, csum, ccnt);
  k_cluster_fin<<<1, 128, 0, stream>>>(csum, ccnt, q);

  // first_pn (WMMA)
  k_build_f1<<<PB, 256, 0, stream>>>(x, sn, q, minid, bufA);
  conv(0, bufA, bufB, nullptr, 64);  bnrelu(0, bufB, 64);
  conv(1, bufB, bufA, nullptr, 64);  bnrelu(1, bufA, 64);
  conv(2, bufA, bufB, nullptr, 64);  bnrelu(2, bufB, 64);

  k_zero<<<(BB*MM*64 + 255)/256, 256, 0, stream>>>(fm1, BB*MM*64);
  k_clustermax<<<PB, 256, 0, stream>>>(bufB, minid, fm1, 64);
  k_build_f2<<<PB, 256, 0, stream>>>(bufB, fm1, minid, bufA);

  // second_pn (WMMA)
  conv(3, bufA, bufB, nullptr, 128); bnrelu(3, bufB, 128);
  conv(4, bufB, bufA, nullptr, 128); bnrelu(4, bufA, 128);

  k_zero<<<(BB*MM*128 + 255)/256, 256, 0, stream>>>(fm2, BB*MM*128);
  k_clustermax<<<PB, 256, 0, stream>>>(bufA, minid, fm2, 128);

  // small M=16 graph (scalar f32)
  k_knn<<<1, 128, 0, stream>>>(q, knn);
  k_build_g<<<2, 256, 0, stream>>>(q, fm2, knn, Xg);
  k_sgemm<<<(384*256 + 255)/256, 256, 0, stream>>>(Xg, PW(5), nullptr, G1, 384, 131, 256);
  k_sbn<<<256, 256, 0, stream>>>(G1, Pg(5), Pz(5), 384, 256);
  k_sgemm<<<(384*256 + 255)/256, 256, 0, stream>>>(G1, PW(6), nullptr, G2, 384, 256, 256);
  k_sbn<<<256, 256, 0, stream>>>(G2, Pg(6), Pz(6), 384, 256);
  k_sgemm<<<(384*256 + 255)/256, 256, 0, stream>>>(G2, PW(7), nullptr, G1, 384, 256, 256);
  k_sbn<<<256, 256, 0, stream>>>(G1, Pg(7), Pz(7), 384, 256);
  k_gmax<<<(128*256 + 255)/256, 256, 0, stream>>>(G1, gm);
  k_cat512<<<(384*512 + 255)/256, 256, 0, stream>>>(G1, gm, C5a);
  k_sgemm<<<(384*512 + 255)/256, 256, 0, stream>>>(C5a, PW(8), nullptr, C5b, 384, 512, 512);
  k_sbn<<<512, 256, 0, stream>>>(C5b, Pg(8), Pz(8), 384, 512);
  k_sgemm<<<(384*512 + 255)/256, 256, 0, stream>>>(C5b, PW(9), nullptr, C5a, 384, 512, 512);
  k_sbn<<<512, 256, 0, stream>>>(C5a, Pg(9), Pz(9), 384, 512);
  k_knnfeat<<<(128*512 + 255)/256, 256, 0, stream>>>(C5a, kf);
  k_build_nf<<<(128*640 + 255)/256, 256, 0, stream>>>(fm2, kf, nf);
  k_sgemm<<<(128*512 + 255)/256, 256, 0, stream>>>(nf, PW(10), nullptr, y512, 128, 640, 512);
  k_sbn<<<512, 256, 0, stream>>>(y512, Pg(10), Pz(10), 128, 512);
  k_sgemm<<<(128*256 + 255)/256, 256, 0, stream>>>(y512, PW(11), nullptr, y256, 128, 512, 256);
  k_sbn<<<256, 256, 0, stream>>>(y256, Pg(11), Pz(11), 128, 256);
  k_sgemm<<<(128*4 + 255)/256, 256, 0, stream>>>(y256, PW(12), Pb(12), ko, 128, 256, 4);
  k_part_nodes<<<2, 256, 0, stream>>>(ko, q, out);

  // inst branch (WMMA) + coefs
  k_build_hin<<<PB, 256, 0, stream>>>(x, sn, bufB);
  conv(5, bufB, bufA, nullptr, 64);  bnrelu(13, bufA, 64);
  conv(6, bufA, bufB, nullptr, 128); bnrelu(14, bufB, 128);
  conv(7, bufB, bufA, Pb(15), 10);   // bias, no norm, no relu
  k_fill<<<1, 256, 0, stream>>>(out + 384, 80, -3.402823466e38f);
  k_coefmax<<<PB, 256, 0, stream>>>(bufA, out + 384);

  // joint branch (npos = B = 8)
  k_sgemm<<<(8*64 + 255)/256, 256, 0, stream>>>(out + 384, PW(16), nullptr, J1, 8, 10, 64);
  k_sbn<<<64, 256, 0, stream>>>(J1, Pg(16), Pz(16), 8, 64);
  k_sgemm<<<(8*64 + 255)/256, 256, 0, stream>>>(J1, PW(17), nullptr, J2, 8, 64, 64);
  k_sbn<<<64, 256, 0, stream>>>(J2, Pg(17), Pz(17), 8, 64);
  k_sgemm<<<1, 256, 0, stream>>>(J2, PW(18), Pb(18), J3, 8, 64, 6);
  k_jp_out<<<1, 64, 0, stream>>>(J3, out);
}
